// DistanceDecoder_32487132627150
// MI455X (gfx1250) — matrix operand order repaired and checked by reading, not compiled, not run on hardware
//
#include <hip/hip_runtime.h>
#include <hip/hip_bf16.h>
#include <math.h>

typedef __bf16 bf16_t;
typedef __attribute__((ext_vector_type(16))) __bf16 v16bf;
typedef __attribute__((ext_vector_type(8)))  __bf16 v8bf;
typedef __attribute__((ext_vector_type(8)))  float   v8f;

// ---- CDNA5 async global->LDS copy (16 bytes per lane), ASYNCcnt-tracked ----
__device__ __forceinline__ void async_copy16(void* lds_dst, const void* gsrc) {
#if defined(__gfx1250__)
  unsigned lds_off = (unsigned)(uintptr_t)lds_dst;   // low 32 bits = LDS offset
  asm volatile("global_load_async_to_lds_b128 %0, %1, off"
               :: "v"(lds_off), "v"(gsrc) : "memory");
#else
  *(uint4*)lds_dst = *(const uint4*)gsrc;
#endif
}

__device__ __forceinline__ void async_wait0() {
#if defined(__gfx1250__)
  asm volatile("s_wait_asynccnt 0x0" ::: "memory");
#endif
}

// ---------------------------------------------------------------- degree ----
__global__ void k_deg_count(const int* __restrict__ dst, int* __restrict__ deg, int E) {
  int e = blockIdx.x * blockDim.x + threadIdx.x;
  if (e < E) atomicAdd(&deg[dst[e]], 1);
}

__global__ void k_deg_fin(const int* __restrict__ deg, float* __restrict__ dinv,
                          float* __restrict__ dinvf, int n) {
  int i = blockIdx.x * blockDim.x + threadIdx.x;
  if (i < n) {
    float d = (float)deg[i] + 1.0f;
    dinv[i]  = rsqrtf(d);
    dinvf[i] = 1.0f / d;
  }
}

// ------------------------------------------- weight transpose -> bf16 [Nout,Kpad]
__global__ void k_wconv(const float* __restrict__ W, bf16_t* __restrict__ Wt,
                        int K, int Kpad, int Nout) {
  int idx = blockIdx.x * blockDim.x + threadIdx.x;
  if (idx >= Nout * Kpad) return;
  int n = idx / Kpad, k = idx - n * Kpad;
  float v = (k < K) ? W[(size_t)k * Nout + n] : 0.0f;
  Wt[idx] = (bf16_t)v;
}

// ------------------------------------------- log map at mu0, padded to K=96 bf16
__global__ void k_logmap(const float* __restrict__ z, bf16_t* __restrict__ zp, int n) {
  int i = blockIdx.x * blockDim.x + threadIdx.x;
  if (i >= n) return;
  const float* zr = z + (size_t)i * 65;
  bf16_t* o = zp + (size_t)i * 96;
  float alpha = fmaxf(zr[0], 1.0f + 1e-7f);
  float d = acoshf(alpha);
  float coef = d * rsqrtf(alpha * alpha - 1.0f);
  o[0] = (bf16_t)0.0f;
  #pragma unroll 8
  for (int j = 1; j < 65; ++j) o[j] = (bf16_t)(coef * zr[j]);
  #pragma unroll
  for (int j = 65; j < 96; ++j) o[j] = (bf16_t)0.0f;
}

// ------------------------------------------- per-edge dist + sym-norm coefficient
__global__ void k_edge_geom(const float* __restrict__ z, const int* __restrict__ src,
                            const int* __restrict__ dst, const float* __restrict__ dinv,
                            float* __restrict__ dist, float* __restrict__ nrm, int E) {
  int e = blockIdx.x * blockDim.x + threadIdx.x;
  if (e >= E) return;
  int s = src[e], d = dst[e];
  const float* zs = z + (size_t)s * 65;
  const float* zd = z + (size_t)d * 65;
  float inner = -zs[0] * zd[0];
  float acc = 0.0f;
  #pragma unroll 8
  for (int i = 1; i < 65; ++i) acc += zs[i] * zd[i];
  inner += acc;
  float alpha = fmaxf(-inner, 1.0f + 1e-7f);
  dist[e] = -acoshf(alpha);
  nrm[e]  = dinv[s] * dinv[d];
}

// ------------------------------------------- node GEMM: H[M,Nout] = X[M,K]@W
// X row-major bf16 [M,K], Wt row-major bf16 [Nout,K].
// Block = 8 waves; wave w owns 16 rows (mtile = blockIdx*8+w). A is register-
// resident (templated K => constant indices). Weights stream through a 16KB
// LDS chunk (32 columns x K) shared by all 8 waves (async global->LDS).
template<int K>
__global__ void k_gemm_node(const bf16_t* __restrict__ X, const bf16_t* __restrict__ Wt,
                            float* __restrict__ H, int Nout, int Mtiles) {
  constexpr int NK = K / 32;
  const int lane = threadIdx.x & 31;
  const int wave = threadIdx.x >> 5;
  const int r    = lane & 15;
  const int half = lane >> 4;
  const int mtile = blockIdx.x * 8 + wave;
  const bool valid = (mtile < Mtiles);

  v16bf af[NK];
  if (valid) {
    const bf16_t* xrow = X + ((size_t)mtile * 16 + r) * K + half * 16;
    #pragma unroll
    for (int t = 0; t < NK; ++t) {
      v8bf lo = *(const v8bf*)(xrow + t * 32);
      v8bf hi = *(const v8bf*)(xrow + t * 32 + 8);
      #pragma unroll
      for (int i = 0; i < 8; ++i) { af[t][i] = lo[i]; af[t][i + 8] = hi[i]; }
    }
  }

  __shared__ bf16_t wlds[32 * K];
  const int nchunk = Nout >> 5;                 // 32 output columns per chunk
  for (int ch = 0; ch < nchunk; ++ch) {
    {
      const bf16_t* g = Wt + (size_t)ch * 32 * K;
      const int nvec = (32 * K) >> 3;           // 16B vectors
      for (int i = threadIdx.x; i < nvec; i += 256)
        async_copy16(wlds + (i << 3), g + (i << 3));
      async_wait0();
    }
    __syncthreads();
    if (valid) {
      #pragma unroll
      for (int sub = 0; sub < 2; ++sub) {
        const bf16_t* wrow = wlds + (sub * 16 + r) * K + half * 16;
        v8f c = {};
        #pragma unroll
        for (int t = 0; t < NK; ++t) {
          v16bf b = *(const v16bf*)(wrow + t * 32);
          c = __builtin_amdgcn_wmma_f32_16x16x32_bf16(false, af[t], false, b,
                                                      (short)0, c, false, false);
        }
        float* out = H + ((size_t)mtile * 16 + half * 8) * Nout + (ch * 2 + sub) * 16 + r;
        #pragma unroll
        for (int i = 0; i < 8; ++i) out[(size_t)i * Nout] = c[i];
      }
    }
    __syncthreads();
  }
}

// ------------------------------------------- out = h*deg^-1 + b (self loop + bias)
__global__ void k_init(const float* __restrict__ h, const float* __restrict__ dinvf,
                       const float* __restrict__ b, float* __restrict__ out, int total, int F) {
  int idx = blockIdx.x * blockDim.x + threadIdx.x;
  if (idx >= total) return;
  int n = idx / F, f = idx - n * F;
  out[idx] = h[idx] * dinvf[n] + b[f];
}

// ------------------------------------------- out[dst] += h[src]*norm (atomic f32)
__global__ void k_scatter(const float* __restrict__ h, const int* __restrict__ src,
                          const int* __restrict__ dst, const float* __restrict__ nrm,
                          float* __restrict__ out, int F, int E) {
  int qpr = F >> 2;
  int idx = blockIdx.x * blockDim.x + threadIdx.x;
  if (idx >= E * qpr) return;
  int e = idx / qpr, q = (idx - e * qpr) << 2;
  int s = src[e], d = dst[e];
  float w = nrm[e];
  const float4 hv = *(const float4*)(h + (size_t)s * F + q);
  float* o = out + (size_t)d * F + q;
  __hip_atomic_fetch_add(o + 0, hv.x * w, __ATOMIC_RELAXED, __HIP_MEMORY_SCOPE_AGENT);
  __hip_atomic_fetch_add(o + 1, hv.y * w, __ATOMIC_RELAXED, __HIP_MEMORY_SCOPE_AGENT);
  __hip_atomic_fetch_add(o + 2, hv.z * w, __ATOMIC_RELAXED, __HIP_MEMORY_SCOPE_AGENT);
  __hip_atomic_fetch_add(o + 3, hv.w * w, __ATOMIC_RELAXED, __HIP_MEMORY_SCOPE_AGENT);
}

// ------------------------------------------- f32 -> bf16 (optional relu)
__global__ void k_tobf(const float* __restrict__ x, bf16_t* __restrict__ y, int n, int relu) {
  int i = blockIdx.x * blockDim.x + threadIdx.x;
  if (i >= n) return;
  float v = x[i];
  if (relu) v = fmaxf(v, 0.0f);
  y[i] = (bf16_t)v;
}

// ------------------------------------------- fused edge MLP heads (r and t)
// Wave handles 16 edges; A = [h4[src] | h4[dst]] bf16 (K=256), register resident.
// Weights for both branches stream through one 16KB LDS chunk (32 cols x 256),
// shared by the 8 waves (128 edges) of the block -> 8x L2-traffic reduction.
__global__ void k_edge_mlp(const bf16_t* __restrict__ h4,
                           const int* __restrict__ src, const int* __restrict__ dst,
                           const bf16_t* __restrict__ rW1t, const float* __restrict__ rb1,
                           const float* __restrict__ rW2, const float* __restrict__ rb2,
                           const bf16_t* __restrict__ tW1t, const float* __restrict__ tb1,
                           const float* __restrict__ tW2, const float* __restrict__ tb2,
                           const float* __restrict__ dist, float* __restrict__ out) {
  const int lane = threadIdx.x & 31;
  const int wave = threadIdx.x >> 5;
  const int r    = lane & 15;
  const int half = lane >> 4;
  const int e16  = (blockIdx.x * 8 + wave) * 16;
  const int s = src[e16 + r], d = dst[e16 + r];

  v16bf af[8];
  #pragma unroll
  for (int t = 0; t < 8; ++t) {
    const int k0 = t * 32 + half * 16;
    const int k1 = t * 32 + half * 16 + 8;
    const bf16_t* p0 = h4 + (size_t)((k0 < 128) ? s : d) * 128 + (k0 & 127);
    const bf16_t* p1 = h4 + (size_t)((k1 < 128) ? s : d) * 128 + (k1 & 127);
    v8bf lo = *(const v8bf*)p0;
    v8bf hi = *(const v8bf*)p1;
    #pragma unroll
    for (int i = 0; i < 8; ++i) { af[t][i] = lo[i]; af[t][i + 8] = hi[i]; }
  }

  __shared__ bf16_t wlds[32 * 256];
  float acc[2][8];
  #pragma unroll
  for (int br = 0; br < 2; ++br)
    #pragma unroll
    for (int i = 0; i < 8; ++i) acc[br][i] = 0.0f;

  #pragma unroll
  for (int br = 0; br < 2; ++br) {
    const bf16_t* Wt = br ? tW1t : rW1t;
    const float*  b1 = br ? tb1  : rb1;
    const float*  w2 = br ? tW2  : rW2;
    for (int ch = 0; ch < 8; ++ch) {           // 8 chunks x 32 columns = 256
      {
        const bf16_t* g = Wt + (size_t)ch * 32 * 256;
        const int nvec = (32 * 256) >> 3;
        for (int i = threadIdx.x; i < nvec; i += 256)
          async_copy16(wlds + (i << 3), g + (i << 3));
        async_wait0();
      }
      __syncthreads();
      #pragma unroll
      for (int sub = 0; sub < 2; ++sub) {
        const int n = (ch * 2 + sub) * 16 + r;
        const bf16_t* wrow = wlds + (sub * 16 + r) * 256 + half * 16;
        v8f c = {};
        #pragma unroll
        for (int t = 0; t < 8; ++t) {
          v16bf b = *(const v16bf*)(wrow + t * 32);
          c = __builtin_amdgcn_wmma_f32_16x16x32_bf16(false, af[t], false, b,
                                                      (short)0, c, false, false);
        }
        float bias = b1[n], w = w2[n];
        #pragma unroll
        for (int i = 0; i < 8; ++i) {
          float v = c[i] + bias;
          v = (v > 0.0f) ? v : 0.2f * v;       // leaky_relu(0.2)
          acc[br][i] += v * w;                 // rank-1 contraction (rW2 / tW2)
        }
      }
      __syncthreads();
    }
  }

  #pragma unroll
  for (int br = 0; br < 2; ++br)
    #pragma unroll
    for (int i = 0; i < 8; ++i)
      for (int m = 1; m < 16; m <<= 1) acc[br][i] += __shfl_xor(acc[br][i], m, 32);

  if (r == 0) {
    const float rb2v = rb2[0], tb2v = tb2[0];
    #pragma unroll
    for (int i = 0; i < 8; ++i) {
      const int e = e16 + half * 8 + i;
      float rv = acc[0][i] + rb2v;
      float tv = acc[1][i] + tb2v;
      float x  = (dist[e] - rv) / tv;
      out[e] = 1.0f / (1.0f + __expf(-x));
    }
  }
}

// ---------------------------------------------------------------------------
extern "C" void kernel_launch(void* const* d_in, const int* in_sizes, int n_in,
                              void* d_out, int out_size, void* d_ws, size_t ws_size,
                              hipStream_t stream) {
  const float* z   = (const float*)d_in[0];
  const int*   ei  = (const int*)d_in[1];
  const float* W1  = (const float*)d_in[2];  const float* b1 = (const float*)d_in[3];
  const float* W2  = (const float*)d_in[4];  const float* b2 = (const float*)d_in[5];
  const float* W3  = (const float*)d_in[6];  const float* b3 = (const float*)d_in[7];
  const float* W4  = (const float*)d_in[8];  const float* b4 = (const float*)d_in[9];
  const float* rW1 = (const float*)d_in[10]; const float* rb1 = (const float*)d_in[11];
  const float* rW2 = (const float*)d_in[12]; const float* rb2 = (const float*)d_in[13];
  const float* tW1 = (const float*)d_in[14]; const float* tb1 = (const float*)d_in[15];
  const float* tW2 = (const float*)d_in[16]; const float* tb2 = (const float*)d_in[17];

  const int N = in_sizes[0] / 65;      // 50000
  const int E = in_sizes[1] / 2;       // 800000
  const int* src = ei;
  const int* dst = ei + E;
  float* out = (float*)d_out;

  // ---- workspace layout ----
  char* ws = (char*)d_ws;
  size_t off = 0;
  auto carve = [&](size_t bytes) -> char* {
    char* p = ws + off;
    off += (bytes + 255) & ~(size_t)255;
    return p;
  };
  bf16_t* zpad  = (bf16_t*)carve((size_t)N * 96 * 2);
  bf16_t* xbf   = (bf16_t*)carve((size_t)N * 256 * 2);
  float*  hA    = (float*)carve((size_t)N * 256 * 4);
  float*  hB    = (float*)carve((size_t)N * 256 * 4);
  bf16_t* h4    = (bf16_t*)carve((size_t)N * 128 * 2);
  int*    degi  = (int*)carve((size_t)N * 4);
  float*  dinv  = (float*)carve((size_t)N * 4);
  float*  dinvf = (float*)carve((size_t)N * 4);
  float*  dist  = (float*)carve((size_t)E * 4);
  float*  nrm   = (float*)carve((size_t)E * 4);
  bf16_t* w1t   = (bf16_t*)carve((size_t)256 * 96 * 2);
  bf16_t* w2t   = (bf16_t*)carve((size_t)256 * 256 * 2);
  bf16_t* w3t   = (bf16_t*)carve((size_t)256 * 256 * 2);
  bf16_t* w4t   = (bf16_t*)carve((size_t)128 * 256 * 2);
  bf16_t* rw1t  = (bf16_t*)carve((size_t)256 * 256 * 2);
  bf16_t* tw1t  = (bf16_t*)carve((size_t)256 * 256 * 2);

  const int TB = 256;
  auto grid1 = [&](long long n) { return (unsigned)((n + TB - 1) / TB); };

  // degrees
  hipMemsetAsync(degi, 0, (size_t)N * 4, stream);
  k_deg_count<<<grid1(E), TB, 0, stream>>>(dst, degi, E);
  k_deg_fin<<<grid1(N), TB, 0, stream>>>(degi, dinv, dinvf, N);

  // weight prep (transpose + bf16)
  k_wconv<<<grid1(256 * 96),  TB, 0, stream>>>(W1,  w1t,  65,  96,  256);
  k_wconv<<<grid1(256 * 256), TB, 0, stream>>>(W2,  w2t,  256, 256, 256);
  k_wconv<<<grid1(256 * 256), TB, 0, stream>>>(W3,  w3t,  256, 256, 256);
  k_wconv<<<grid1(128 * 256), TB, 0, stream>>>(W4,  w4t,  256, 256, 128);
  k_wconv<<<grid1(256 * 256), TB, 0, stream>>>(rW1, rw1t, 256, 256, 256);
  k_wconv<<<grid1(256 * 256), TB, 0, stream>>>(tW1, tw1t, 256, 256, 256);

  // geometry
  k_logmap<<<grid1(N), TB, 0, stream>>>(z, zpad, N);
  k_edge_geom<<<grid1(E), TB, 0, stream>>>(z, src, dst, dinv, dist, nrm, E);

  const int Mtiles = N / 16;                       // 3125
  const unsigned gblocks = (unsigned)((Mtiles + 7) / 8);

  // ---- layer 1: [N,96] -> [N,256]
  k_gemm_node<96><<<gblocks, TB, 0, stream>>>(zpad, w1t, hA, 256, Mtiles);
  k_init<<<grid1((long long)N * 256), TB, 0, stream>>>(hA, dinvf, b1, hB, N * 256, 256);
  k_scatter<<<grid1((long long)E * 64), TB, 0, stream>>>(hA, src, dst, nrm, hB, 256, E);
  k_tobf<<<grid1((long long)N * 256), TB, 0, stream>>>(hB, xbf, N * 256, 1);

  // ---- layer 2
  k_gemm_node<256><<<gblocks, TB, 0, stream>>>(xbf, w2t, hA, 256, Mtiles);
  k_init<<<grid1((long long)N * 256), TB, 0, stream>>>(hA, dinvf, b2, hB, N * 256, 256);
  k_scatter<<<grid1((long long)E * 64), TB, 0, stream>>>(hA, src, dst, nrm, hB, 256, E);
  k_tobf<<<grid1((long long)N * 256), TB, 0, stream>>>(hB, xbf, N * 256, 1);

  // ---- layer 3
  k_gemm_node<256><<<gblocks, TB, 0, stream>>>(xbf, w3t, hA, 256, Mtiles);
  k_init<<<grid1((long long)N * 256), TB, 0, stream>>>(hA, dinvf, b3, hB, N * 256, 256);
  k_scatter<<<grid1((long long)E * 64), TB, 0, stream>>>(hA, src, dst, nrm, hB, 256, E);
  k_tobf<<<grid1((long long)N * 256), TB, 0, stream>>>(hB, xbf, N * 256, 1);

  // ---- layer 4: [N,256] -> [N,128], no relu
  k_gemm_node<256><<<gblocks, TB, 0, stream>>>(xbf, w4t, hA, 128, Mtiles);
  k_init<<<grid1((long long)N * 128), TB, 0, stream>>>(hA, dinvf, b4, hB, N * 128, 128);
  k_scatter<<<grid1((long long)E * 32), TB, 0, stream>>>(hA, src, dst, nrm, hB, 128, E);
  k_tobf<<<grid1((long long)N * 128), TB, 0, stream>>>(hB, h4, N * 128, 0);

  // ---- fused edge MLP + sigmoid ----
  k_edge_mlp<<<(unsigned)(E / 128), TB, 0, stream>>>(h4, src, dst,
                                                     rw1t, rb1, rW2, rb2,
                                                     tw1t, tb1, tW2, tb2,
                                                     dist, out);
  (void)n_in; (void)out_size; (void)ws_size;
}
